// CorrLSTM_29609504538948
// MI455X (gfx1250) — matrix operand order repaired and checked by reading, not compiled, not run on hardware
//
#include <hip/hip_runtime.h>
#include <hip/hip_bf16.h>

// ---------------- problem constants ----------------
#define P7   7
#define PADR 3
#define PP   49          // correlation channels
#define NCH_IN  98       // 2*PP conv input channels
#define NCH_OUT 196      // 4*PP conv output channels
#define KTOT 882         // 98 * 9
#define NKC  28          // ceil(882/32) K-chunks of 32
#define NT   13          // N tiles of 16 (208 cols, 196 valid)
#define BSTR 24          // bfrag row stride in halves (48 B, 16B-aligned)
#define BATCH 8

typedef __attribute__((ext_vector_type(16))) _Float16 v16h;
typedef __attribute__((ext_vector_type(8)))  _Float16 v8h;
typedef __attribute__((ext_vector_type(8)))  float    v8f;

// =====================================================================
// Stage 1: spatial correlation + leaky ReLU.
// One block = one (batch, spatial tile). Channel-chunked LDS tiling.
// =====================================================================
__global__ __launch_bounds__(256) void corr_kernel(
    const float* __restrict__ x, const float* __restrict__ xp,
    float* __restrict__ corr, int C, int S, int TS) {
  const int tid = threadIdx.x;
  const int tilesPer = S / TS;
  const int bi = blockIdx.x / (tilesPer * tilesPer);
  const int t  = blockIdx.x % (tilesPer * tilesPer);
  const int y0 = (t / tilesPer) * TS;
  const int x0 = (t % tilesPer) * TS;
  const int TA  = TS * TS;          // <= 64
  const int FW  = TS + 6;           // <= 14
  const int F2A = FW * FW;          // <= 196

  __shared__ float f1s[32 * 64];
  __shared__ float f2s[32 * 196];

  const int nout = PP * TA;
  float acc[13];
  int   od[13], oy[13], ox[13];
  int nacc = 0;
  for (int idx = tid; idx < nout; idx += 256) {
    od[nacc] = idx / TA;
    int p = idx % TA;
    oy[nacc] = p / TS; ox[nacc] = p % TS;
    acc[nacc] = 0.f;
    ++nacc;
  }

  for (int c0 = 0; c0 < C; c0 += 32) {
    __syncthreads();
    for (int i = tid; i < 32 * TA; i += 256) {          // f1 tile
      int k = i / TA, p = i % TA;
      int yy = y0 + p / TS, xx = x0 + p % TS;
      f1s[k * TA + p] = x[(((size_t)bi * C + c0 + k) * S + yy) * S + xx];
    }
    for (int i = tid; i < 32 * F2A; i += 256) {         // padded f2 tile
      int k = i / F2A, p = i % F2A;
      int yy = y0 + p / FW - PADR, xx = x0 + p % FW - PADR;
      float v = 0.f;
      if (yy >= 0 && yy < S && xx >= 0 && xx < S)
        v = xp[(((size_t)bi * C + c0 + k) * S + yy) * S + xx];
      f2s[k * F2A + p] = v;
    }
    __syncthreads();
    for (int j = 0; j < nacc; ++j) {
      const int di = od[j] / P7, dj = od[j] % P7;
      const int p1 = oy[j] * TS + ox[j];
      const int p2 = (oy[j] + di) * FW + (ox[j] + dj);
      float a = acc[j];
#pragma unroll 8
      for (int k = 0; k < 32; ++k)
        a += f1s[k * TA + p1] * f2s[k * F2A + p2];
      acc[j] = a;
    }
  }
  for (int j = 0; j < nacc; ++j) {
    float v = acc[j];
    v = v > 0.f ? v : 0.01f * v;                         // leaky ReLU
    corr[(((size_t)bi * PP + od[j]) * S + (y0 + oy[j])) * S + (x0 + ox[j])] = v;
  }
}

// =====================================================================
// Stage 2: implicit-GEMM 3x3 conv (WMMA f16 -> f32) + LSTM gates.
// Block = 128 threads = 4 waves, each wave owns one M-tile (16 rows),
// all 13 N-tiles. Weights staged f32->f16 in LDS pre-swizzled into the
// WMMA B-fragment layout. B fragments are consumed in groups of 4 with
// a sched_barrier between the 8 ds_load_b128 and the 4 WMMAs so one
// LDS round-trip wait is amortized over 4 matrix ops.
// =====================================================================
__global__ __launch_bounds__(128) void convlstm_kernel(
    const float* __restrict__ corr, const float* __restrict__ hin,
    const float* __restrict__ cin,  const float* __restrict__ w,
    const float* __restrict__ bias, float* __restrict__ hout,
    float* __restrict__ cout, int S, int Mtotal) {
  // bfrag[nt][lane][j] = B[kbase + ((lane&16)>>1) + j + (j&8), nt*16 + (lane&15)]
  __shared__ _Float16 bfrag[NT][32][BSTR];
  __shared__ float    ep[4][16][208];    // per-wave epilogue staging

  const int tid  = threadIdx.x;
  const int lane = tid & 31;
  const int wv   = tid >> 5;
  const int hi   = lane >> 4;            // half-wave select
  const int mrow = blockIdx.x * 64 + wv * 16 + (lane & 15);
  const bool mvalid = mrow < Mtotal;
  int pb = 0, py = 0, px = 0;
  if (mvalid) {
    pb = mrow / (S * S);
    int rem = mrow % (S * S);
    py = rem / S; px = rem % S;
  }

  v8f acc[NT];
#pragma unroll
  for (int n = 0; n < NT; ++n)
#pragma unroll
    for (int e = 0; e < 8; ++e) acc[n][e] = 0.f;

  const _Float16* bp = &bfrag[0][lane][0];   // one LDS base; tiles via imm offset
  auto loadB = [&](int n) -> v16h {
    v8h lo = *(const v8h*)(bp + n * 32 * BSTR);
    v8h hh = *(const v8h*)(bp + n * 32 * BSTR + 8);
    return __builtin_shufflevector(lo, hh, 0, 1, 2, 3, 4, 5, 6, 7,
                                   8, 9, 10, 11, 12, 13, 14, 15);
  };

  for (int kc = 0; kc < NKC; ++kc) {
    const int kbase = kc * 32;

    // ---- build im2col A fragment first (global loads issue early) ----
    // lane<16 holds K = {0..7, 16..23}; lane>=16 holds K = {8..15, 24..31}
    v16h a;
#pragma unroll
    for (int vv = 0; vv < 8; ++vv) {
      int kk = ((vv < 4) ? 2 * vv : 16 + 2 * (vv - 4)) + hi * 8;
#pragma unroll
      for (int e = 0; e < 2; ++e) {
        int k = kbase + kk + e;
        float val = 0.f;
        if (mvalid && k < KTOT) {
          int ic = k / 9, r = k % 9;
          int yy = py + r / 3 - 1, xx = px + r % 3 - 1;
          if (yy >= 0 && yy < S && xx >= 0 && xx < S) {
            val = (ic < PP)
              ? corr[(((size_t)pb * PP + ic) * S + yy) * S + xx]
              : hin[(((size_t)pb * PP + (ic - PP)) * S + yy) * S + xx];
          }
        }
        a[2 * vv + e] = (_Float16)val;
      }
    }

    // ---- stage weight chunk into B-fragment layout ----
    __syncthreads();                       // previous chunk done reading bfrag
    for (int i = tid; i < NT * 32 * 16; i += 128) {
      int nt  = i >> 9;                    // /512
      int rem = i & 511;
      int L   = rem >> 4;
      int j   = rem & 15;
      int n   = nt * 16 + (L & 15);
      int k   = kbase + ((L & 16) >> 1) + j + (j & 8);
      float v = 0.f;
      if (n < NCH_OUT && k < KTOT) {
        int ic = k / 9, r = k % 9;
        v = w[((n * NCH_IN + ic) * 3 + (r / 3)) * 3 + (r % 3)];
        int k2 = k + 32;                   // warm next chunk (L2-resident)
        if (k2 < KTOT) {
          int ic2 = k2 / 9, r2 = k2 % 9;
          __builtin_prefetch(&w[((n * NCH_IN + ic2) * 3 + (r2 / 3)) * 3 + (r2 % 3)], 0, 1);
        }
      }
      bfrag[nt][L][j] = (_Float16)v;
    }
    __syncthreads();

    // ---- 13 WMMAs in groups of 4: loads batched ahead of matrix ops ----
#pragma unroll
    for (int g = 0; g < NT; g += 4) {
      constexpr int GMAX = 4;
      v16h b0, b1, b2, b3;
      const int remn = NT - g;             // compile-time under full unroll
      b0 = loadB(g);
      if (remn > 1) b1 = loadB(g + 1);
      if (remn > 2) b2 = loadB(g + 2);
      if (remn > 3) b3 = loadB(g + 3);
      __builtin_amdgcn_sched_barrier(0);   // keep loads above the WMMAs
      acc[g] = __builtin_amdgcn_wmma_f32_16x16x32_f16(
          false, a, false, b0, (short)0, acc[g], false, false);
      if (remn > 1)
        acc[g + 1] = __builtin_amdgcn_wmma_f32_16x16x32_f16(
            false, a, false, b1, (short)0, acc[g + 1], false, false);
      if (remn > 2)
        acc[g + 2] = __builtin_amdgcn_wmma_f32_16x16x32_f16(
            false, a, false, b2, (short)0, acc[g + 2], false, false);
      if (remn > 3)
        acc[g + 3] = __builtin_amdgcn_wmma_f32_16x16x32_f16(
            false, a, false, b3, (short)0, acc[g + 3], false, false);
      (void)GMAX;
    }
  }

  __syncthreads();
  // C layout: VGPR r -> row r + 8*hi, col = lane%16
#pragma unroll
  for (int n = 0; n < NT; ++n)
#pragma unroll
    for (int r = 0; r < 8; ++r)
      ep[wv][r + hi * 8][n * 16 + (lane & 15)] = acc[n][r];
  __syncthreads();

  // LSTM epilogue: 16 rows x 49 corr-channels per wave
  for (int t = lane; t < 16 * PP; t += 32) {
    int mr = t / PP, q = t % PP;
    int m = blockIdx.x * 64 + wv * 16 + mr;
    if (m >= Mtotal) continue;
    int b   = m / (S * S);
    int rem = m % (S * S), y = rem / S, xq = rem % S;
    float ci = ep[wv][mr][q]          + bias[q];
    float cf = ep[wv][mr][PP + q]     + bias[PP + q];
    float co = ep[wv][mr][2 * PP + q] + bias[2 * PP + q];
    float cg = ep[wv][mr][3 * PP + q] + bias[3 * PP + q];
    float ig = 1.f / (1.f + __expf(-ci));
    float fg = 1.f / (1.f + __expf(-cf));
    float og = 1.f / (1.f + __expf(-co));
    float gg = tanhf(cg);
    size_t off = (((size_t)b * PP + q) * S + y) * S + xq;
    float cn = fg * cin[off] + ig * gg;
    hout[off] = og * tanhf(cn);
    cout[off] = cn;
  }
}

// =====================================================================
// Host launcher
// =====================================================================
extern "C" void kernel_launch(void* const* d_in, const int* in_sizes, int n_in,
                              void* d_out, int out_size, void* d_ws, size_t ws_size,
                              hipStream_t stream) {
  static const int CHANS[6] = {512, 1024, 512, 256, 256, 256};
  static const int HWs[6]   = {64, 32, 16, 8, 4, 2};
  (void)in_sizes; (void)n_in; (void)out_size; (void)d_ws; (void)ws_size;

  float* out = (float*)d_out;
  size_t hoff[6];
  size_t total = 0;
  for (int l = 0; l < 6; ++l) {
    hoff[l] = total;
    total += (size_t)BATCH * PP * HWs[l] * HWs[l];
  }
  // d_out layout: [h0..h5][c0..c5][corr0..corr5]

  // Stage 1: correlations (write final corr outputs)
  for (int l = 0; l < 6; ++l) {
    const int S = HWs[l], C = CHANS[l];
    const int TS = (S >= 8) ? 8 : S;
    const int tiles = (S / TS) * (S / TS);
    const float* x  = (const float*)d_in[6 * l + 0];
    const float* xp = (const float*)d_in[6 * l + 1];
    float* corr = out + 2 * total + hoff[l];
    corr_kernel<<<dim3(BATCH * tiles), dim3(256), 0, stream>>>(x, xp, corr, C, S, TS);
  }

  // Stage 2: conv (WMMA) + LSTM (reads corr section of d_out)
  for (int l = 0; l < 6; ++l) {
    const int S = HWs[l];
    const int Mtotal = BATCH * S * S;
    const int grid = (Mtotal + 63) / 64;
    const float* hin  = (const float*)d_in[6 * l + 2];
    const float* cin  = (const float*)d_in[6 * l + 3];
    const float* w    = (const float*)d_in[6 * l + 4];
    const float* bias = (const float*)d_in[6 * l + 5];
    const float* corr = out + 2 * total + hoff[l];
    float* hout = out + hoff[l];
    float* cout = out + total + hoff[l];
    convlstm_kernel<<<dim3(grid), dim3(128), 0, stream>>>(
        corr, hin, cin, w, bias, hout, cout, S, Mtotal);
  }
}
